// FuY_13331578487103
// MI455X (gfx1250) — matrix operand order repaired and checked by reading, not compiled, not run on hardware
//
#include <hip/hip_runtime.h>
#include <hip/hip_bf16.h>

typedef __attribute__((ext_vector_type(16))) _Float16 v16h;
typedef __attribute__((ext_vector_type(8)))  _Float16 v8h;
typedef __attribute__((ext_vector_type(8)))  float    v8f;

#define T_LEN   2048
#define NB      16
#define ND      64
#define NSER    (NB * ND)          // 1024 series
#define FPAD    1024               // rows r = 0..1023 <-> freq k = r (r=0 masked later)
#define PRED    720
#define TOUT    (T_LEN + PRED)     // 2768
#define KTOP    16
#define TWO_PI_OVER_T 3.0679615757712823e-3f  // 2*pi/2048

// ---------------------------------------------------------------------------
// Kernel 1: basis tables, f16.  bc[r][n] = cos(2*pi*r*n/T), bs[r][n] = -sin(..)
// Row-major [freq][time]: time (the GEMM K dim) is contiguous -> A fragments
// load as b128.  Angle reduced exactly via (r*n) mod 2048.
// ---------------------------------------------------------------------------
__global__ void fits_build_basis(_Float16* __restrict__ bc, _Float16* __restrict__ bs) {
    int idx = blockIdx.x * blockDim.x + threadIdx.x;   // over FPAD*T_LEN
    if (idx >= FPAD * T_LEN) return;
    int r = idx >> 11;
    int n = idx & (T_LEN - 1);
    int m = (r * n) & (T_LEN - 1);
    float th = (float)m * TWO_PI_OVER_T;
    bc[idx] = (_Float16)__cosf(th);
    bs[idx] = (_Float16)(-__sinf(th));
}

// ---------------------------------------------------------------------------
// Kernel 2: x (b,t,d) f32  ->  xh[series][time] f16  (series s = b*64 + d)
// LDS-tiled transpose: coalesced f32 reads along d, contiguous f16 writes
// along t.  Time-contiguous rows make B fragments pure b128 loads.
// ---------------------------------------------------------------------------
__global__ __launch_bounds__(256) void fits_pack_x(const float* __restrict__ x,
                                                   _Float16* __restrict__ xh) {
    __shared__ _Float16 tile[16][ND + 8];
    const int b   = blockIdx.y;
    const int n0  = blockIdx.x * 16;
    const int tid = threadIdx.x;

    for (int r = tid >> 6; r < 16; r += 4) {
        const int d = tid & 63;
        tile[r][d] = (_Float16)x[((size_t)(b * T_LEN + n0 + r)) * ND + d];
    }
    __syncthreads();

    const int nn = tid & 15;
    for (int dbase = 0; dbase < ND; dbase += 16) {
        const int d = dbase + (tid >> 4);
        xh[((size_t)(b * ND + d)) * T_LEN + n0 + nn] = tile[nn][d];
    }
}

// ---------------------------------------------------------------------------
// Kernel 3: DFT as GEMM via WMMA.
//   Re[r][s] = sum_n bc[r][n]*xh[s][n];  Im[r][s] = sum_n bs[r][n]*xh[s][n]
// Block = 8 waves.  Wave w handles freq tile (bx*8+w) x 4 series tiles
// (series group by), so each A fragment feeds 4 WMMAs.  The 4 KB B panel per
// k-step is staged once into LDS by the whole block and read by all 8 waves.
// Fragment layouts used here (wave32):
//   A 16x32 f16: lane m=L&15 -> row M; halves = rowC[hi*16 .. +7] ++
//                rowC[hi*16+8 .. +7]  (two b128 loads)
//   B 32x16 f16: lane -> col N=L&15; halves = col[hi*16 .. +15] (contiguous)
//   C/D 16x16 f32: lane n=L&15, VGPR v -> row M=(L>>4)*8+v
// ---------------------------------------------------------------------------
__global__ __launch_bounds__(256) void fits_dft_wmma(
        const _Float16* __restrict__ bc, const _Float16* __restrict__ bs,
        const _Float16* __restrict__ xh,
        float* __restrict__ Re, float* __restrict__ Im) {
    __shared__ __attribute__((aligned(32))) _Float16 sB[4 * 16 * 32]; // [tile][n][k]

    const int wave = threadIdx.x >> 5;
    const int lane = threadIdx.x & 31;
    const int m    = lane & 15;
    const int hi   = lane >> 4;
    const int fr   = blockIdx.x * 8 + wave;   // freq tile 0..63
    const int sg   = blockIdx.y;              // series group: tiles sg*4 .. sg*4+3

    v8f accRe[4] = {};
    v8f accIm[4] = {};

    const _Float16* rowC = bc + (size_t)(fr * 16 + m) * T_LEN;
    const _Float16* rowS = bs + (size_t)(fr * 16 + m) * T_LEN;

    // cooperative staging assignment: 256 threads x 16B = 4KB panel
    const int st_t = threadIdx.x >> 6;          // series tile 0..3
    const int st_n = threadIdx.x & 15;          // column within tile
    const int st_k = (threadIdx.x >> 4) & 3;    // 8-half chunk 0..3
    const _Float16* srcB = xh + (size_t)(sg * 64 + st_t * 16 + st_n) * T_LEN + st_k * 8;
    _Float16*       dstB = sB + st_t * 512 + st_n * 32 + st_k * 8;

    for (int kk = 0; kk < T_LEN; kk += 32) {
        __syncthreads();
        *(v8h*)dstB = *(const v8h*)(srcB + kk);
        __syncthreads();

        const v8h a0 = *(const v8h*)(rowC + kk + hi * 16);
        const v8h a1 = *(const v8h*)(rowC + kk + hi * 16 + 8);
        const v8h s0 = *(const v8h*)(rowS + kk + hi * 16);
        const v8h s1 = *(const v8h*)(rowS + kk + hi * 16 + 8);
        const v16h aC = __builtin_shufflevector(a0, a1, 0, 1, 2, 3, 4, 5, 6, 7,
                                                8, 9, 10, 11, 12, 13, 14, 15);
        const v16h aS = __builtin_shufflevector(s0, s1, 0, 1, 2, 3, 4, 5, 6, 7,
                                                8, 9, 10, 11, 12, 13, 14, 15);
#pragma unroll
        for (int t = 0; t < 4; ++t) {
            const v16h bX = *(const v16h*)(sB + t * 512 + m * 32 + hi * 16);
            accRe[t] = __builtin_amdgcn_wmma_f32_16x16x32_f16(
                false, aC, false, bX, (short)0, accRe[t], false, false);
            accIm[t] = __builtin_amdgcn_wmma_f32_16x16x32_f16(
                false, aS, false, bX, (short)0, accIm[t], false, false);
        }
    }

    const int mb = hi * 8;
#pragma unroll
    for (int t = 0; t < 4; ++t) {
#pragma unroll
        for (int v = 0; v < 8; ++v) {
            const size_t o = (size_t)(fr * 16 + mb + v) * NSER + sg * 64 + t * 16 + m;
            Re[o] = accRe[t][v];
            Im[o] = accIm[t][v];
        }
    }
}

// ---------------------------------------------------------------------------
// Kernel 4: per-series top-16 by |X|^2 (DC row masked; Nyquist never computed),
// emit synthesis coefficients (k, 2*Re/T, -2*Im/T).  One block per series.
// ---------------------------------------------------------------------------
__global__ __launch_bounds__(128) void fits_topk(
        const float* __restrict__ Re, const float* __restrict__ Im,
        int* __restrict__ kIdx, float* __restrict__ ca, float* __restrict__ cb) {
    __shared__ float mag[FPAD];
    __shared__ float redV[128];
    __shared__ int   redI[128];
    const int s = blockIdx.x;
    const int t = threadIdx.x;

    for (int r = t; r < FPAD; r += 128) {
        float re = Re[(size_t)r * NSER + s];
        float im = Im[(size_t)r * NSER + s];
        mag[r] = (r == 0) ? -1.0f : re * re + im * im;   // mask DC
    }
    __syncthreads();

    for (int j = 0; j < KTOP; ++j) {
        float bv = -2.0f; int bi = 0;
        for (int r = t; r < FPAD; r += 128) {
            float v = mag[r];
            if (v > bv) { bv = v; bi = r; }
        }
        redV[t] = bv; redI[t] = bi;
        __syncthreads();
        for (int off = 64; off > 0; off >>= 1) {
            if (t < off && redV[t + off] > redV[t]) {
                redV[t] = redV[t + off]; redI[t] = redI[t + off];
            }
            __syncthreads();
        }
        const int r = redI[0];
        if (t == 0) {
            float re = Re[(size_t)r * NSER + s];
            float im = Im[(size_t)r * NSER + s];
            kIdx[s * KTOP + j] = r;
            ca[s * KTOP + j]   =  2.0f * re / (float)T_LEN;
            cb[s * KTOP + j]   = -2.0f * im / (float)T_LEN;
            mag[r] = -1.0f;
        }
        __syncthreads();
    }
}

// ---------------------------------------------------------------------------
// Kernel 5: sparse synthesis.  Block = 64 channels x 4 time rows for one batch;
// coefficients for the 64 series staged in LDS.  Exact integer angle reduction.
// ---------------------------------------------------------------------------
__global__ __launch_bounds__(256) void fits_synth(
        const int* __restrict__ kIdx, const float* __restrict__ ca,
        const float* __restrict__ cb, float* __restrict__ out) {
    __shared__ int   sk[ND * KTOP];
    __shared__ float sa[ND * KTOP];
    __shared__ float sb[ND * KTOP];
    const int b  = blockIdx.y;
    const int d  = threadIdx.x & (ND - 1);
    const int nr = threadIdx.x >> 6;
    const int n  = blockIdx.x * 4 + nr;

    for (int i = threadIdx.x; i < ND * KTOP; i += 256) {
        const int s = b * ND + (i >> 4);
        const int j = i & (KTOP - 1);
        sk[i] = kIdx[s * KTOP + j];
        sa[i] = ca[s * KTOP + j];
        sb[i] = cb[s * KTOP + j];
    }
    __syncthreads();
    if (n >= TOUT) return;

    float acc = 0.0f;
#pragma unroll
    for (int j = 0; j < KTOP; ++j) {
        const int   i  = d * KTOP + j;
        const int   mm = (sk[i] * n) & (T_LEN - 1);
        const float th = (float)mm * TWO_PI_OVER_T;
        acc += sa[i] * __cosf(th) + sb[i] * __sinf(th);
    }
    out[((size_t)(b * TOUT + n)) * ND + d] = acc;
}

// ---------------------------------------------------------------------------
// Launch.  Workspace layout (bytes):
//   [0,4M)   basis cos f16   [4M,8M)  basis -sin f16   [8M,12M) xh f16
//   [12M,16M) Re f32         [16M,20M) Im f32
//   [20M, +64K) kIdx  [+64K..+128K) ca  [+128K..+192K) cb     (~20.2 MB total)
// ---------------------------------------------------------------------------
extern "C" void kernel_launch(void* const* d_in, const int* in_sizes, int n_in,
                              void* d_out, int out_size, void* d_ws, size_t ws_size,
                              hipStream_t stream) {
    (void)in_sizes; (void)n_in; (void)out_size; (void)ws_size;
    const float* x = (const float*)d_in[0];
    char* ws = (char*)d_ws;

    _Float16* bc = (_Float16*)(ws);
    _Float16* bs = (_Float16*)(ws + ((size_t)4  << 20));
    _Float16* xh = (_Float16*)(ws + ((size_t)8  << 20));
    float*    Re = (float*)   (ws + ((size_t)12 << 20));
    float*    Im = (float*)   (ws + ((size_t)16 << 20));
    int*    kIdx = (int*)     (ws + ((size_t)20 << 20));
    float*    ca = (float*)   (ws + ((size_t)20 << 20) + (64u << 10));
    float*    cb = (float*)   (ws + ((size_t)20 << 20) + (128u << 10));

    fits_build_basis<<<(FPAD * T_LEN + 255) / 256, 256, 0, stream>>>(bc, bs);
    fits_pack_x<<<dim3(T_LEN / 16, NB), 256, 0, stream>>>(x, xh);
    fits_dft_wmma<<<dim3(8, 16), 256, 0, stream>>>(bc, bs, xh, Re, Im);
    fits_topk<<<NSER, 128, 0, stream>>>(Re, Im, kIdx, ca, cb);
    fits_synth<<<dim3(TOUT / 4, NB), 256, 0, stream>>>(kIdx, ca, cb, (float*)d_out);
}
